// FreqGuidedCrossAttn_47974784696768
// MI455X (gfx1250) — matrix operand order repaired and checked by reading, not compiled, not run on hardware
//
#include <hip/hip_runtime.h>
#include <math.h>

typedef __attribute__((ext_vector_type(16))) _Float16 v16h;
typedef __attribute__((ext_vector_type(8)))  _Float16 v8h;
typedef __attribute__((ext_vector_type(8)))  float    v8f;
typedef unsigned int v4u __attribute__((ext_vector_type(4)));
typedef int v4i __attribute__((ext_vector_type(4)));
typedef int v8i __attribute__((ext_vector_type(8)));

#ifndef __has_builtin
#define __has_builtin(x) 0
#endif
#if defined(__HIP_DEVICE_COMPILE__) && __has_builtin(__builtin_amdgcn_tensor_load_to_lds)
#define USE_TDM 1
#else
#define USE_TDM 0
#endif

// ---------------- problem dims ----------------
#define BB    2
#define CC    32
#define DD    64
#define SS    (64*64*64)
#define WFH   33          // rfft half-spectrum width
#define HIDD  128

// ---------------- workspace layout (bytes) ----------------
#define OFF_FFT 0ull
#define SZ_FFT  ((unsigned long long)BB*CC*DD*DD*WFH*8ull)   // float2 spectrum
#define OFF_NUM (OFF_FFT + SZ_FFT)                           // [B][K][C] f32 band sums
#define OFF_DEN (OFF_NUM + 1024ull)                          // [K] f32 mask counts
#define OFF_KT  (OFF_DEN + 256ull)                           // [B][4][128] f32 k tokens
#define OFF_VT  (OFF_KT  + 4096ull)                          // [B][4][128] f32 v tokens
#define OFF_WT  (OFF_VT  + 4096ull)                          // f16 transposed weights

// WT sub-offsets (in halfs), each stored [N][K] (transposed, f16)
#define WT_SP  0
#define WT_Q   (WT_SP + 128*32)
#define WT_O   (WT_Q  + 128*128)
#define WT_M1  (WT_O  + 128*128)
#define WT_M2  (WT_M1 + 256*128)
#define WT_OP  (WT_M2 + 128*256)
#define WT_R   (WT_OP + 64*128)
#define WT_TOT (WT_R  + 64*32)          // 112640 halfs = 225280 B = 28160 x 8B

// LDS float-constant vector offsets (floats)
#define CV_SPB 0
#define CV_QB  128
#define CV_OB  256
#define CV_M1B 384
#define CV_M2B 640
#define CV_OPB 768
#define CV_RB  832
#define CV_NQG 896
#define CV_NQB 1024
#define CV_NOG 1152
#define CV_NOB 1280
#define CV_KT  1408
#define CV_VT  1920
#define CV_TOT 2432

// main-kernel dynamic LDS: weights + consts + 8x(16x256) stage + 8x(16x32) input stage
#define LDS_MAIN_BYTES (WT_TOT*2 + CV_TOT*4 + 8*16*256*2 + 8*16*32*2)   // 308736 B < 320KB

// =====================================================================
// small helpers
// =====================================================================
__device__ __forceinline__ void lds_fence() {
  // CDNA5 split-counter wait: all prior DS ops complete, compiler memory fence
  asm volatile("s_wait_dscnt 0" ::: "memory");
}

__device__ __forceinline__ float red16(float v) {
  // sum across a 16-lane group (lanes 0-15 / 16-31 stay separate on wave32)
  v += __shfl_xor(v, 1, 32);
  v += __shfl_xor(v, 2, 32);
  v += __shfl_xor(v, 4, 32);
  v += __shfl_xor(v, 8, 32);
  return v;
}

__device__ __forceinline__ v8f wmma16x16x32(v16h a, v16h b, v8f c) {
  return __builtin_amdgcn_wmma_f32_16x16x32_f16(false, a, false, b, (short)0, c, false, false);
}

// A fragment (16x32 f16, row-major rows):
// lane holds row m=lane&15; halfs[0..7]=K kb..kb+7, halfs[8..15]=K 16+kb..23+kb, kb=8*(lane>=16)
__device__ __forceinline__ v16h ld_fragA(const _Float16* p, int kb) {
  union { v16h v; v8h h[2]; } u;
  u.h[0] = *(const v8h*)(p + kb);
  u.h[1] = *(const v8h*)(p + 16 + kb);
  return u.v;
}

// B fragment (32x16 f16) from transposed [N][K] storage:
// lane holds col n=lane&15; halfs[0..7]=K kb..kb+7, halfs[8..15]=K 16+kb..23+kb, kb=8*hi16
__device__ __forceinline__ v16h ld_fragB(const _Float16* rowp, int hi16) {
  union { v16h v; v8h h[2]; } u;
  u.h[0] = *(const v8h*)(rowp + hi16 * 8);
  u.h[1] = *(const v8h*)(rowp + 16 + hi16 * 8);
  return u.v;
}

// generic 16xN = (16xK) @ (KxN) tile GEMM; compile-time bias/accumulate (no branches)
template <int NT, int KC, bool ACCUM, bool BIAS>
__device__ __forceinline__ void gemm_tile(const _Float16* A, int astride,
                                          const _Float16* W, int wstride,
                                          const float* bias, v8f* acc,
                                          int mlane, int kb, int hi16) {
  const _Float16* arow = A + mlane * astride;
#pragma unroll
  for (int nt = 0; nt < NT; ++nt) {
    v8f c;
    if constexpr (ACCUM) {
      c = acc[nt];
    } else if constexpr (BIAS) {
      float bv = bias[nt * 16 + mlane];
#pragma unroll
      for (int i = 0; i < 8; ++i) c[i] = bv;
    } else {
#pragma unroll
      for (int i = 0; i < 8; ++i) c[i] = 0.f;
    }
    const _Float16* wrow = W + (nt * 16 + mlane) * wstride;
#pragma unroll
    for (int kc = 0; kc < KC; ++kc)
      c = wmma16x16x32(ld_fragA(arow + kc * 32, kb), ld_fragB(wrow + kc * 32, hi16), c);
    acc[nt] = c;
  }
}

// dump accumulator tiles to LDS f16 row-major [16][stride]
__device__ __forceinline__ void dump_acc(_Float16* stg, int stride, const v8f* acc,
                                         int ntiles, int mlane, int hi16) {
#pragma unroll
  for (int nt = 0; nt < 8; ++nt) {
    if (nt >= ntiles) break;
#pragma unroll
    for (int r = 0; r < 8; ++r)
      stg[(r + hi16 * 8) * stride + nt * 16 + mlane] = (_Float16)acc[nt][r];
  }
}

// LayerNorm over 128 features (8 tiles) + affine, dumped to LDS f16 [16][stride]
__device__ __forceinline__ void ln_dump(const v8f* acc, const float* g, const float* bt,
                                        _Float16* stg, int stride, int mlane, int hi16) {
  float mean[8], inv[8];
#pragma unroll
  for (int r = 0; r < 8; ++r) {
    float s = 0.f, q = 0.f;
#pragma unroll
    for (int nt = 0; nt < 8; ++nt) { float x = acc[nt][r]; s += x; q += x * x; }
    s = red16(s); q = red16(q);
    float m = s * (1.f / 128.f);
    float v = q * (1.f / 128.f) - m * m;
    mean[r] = m;
    inv[r] = rsqrtf(v + 1e-5f);
  }
#pragma unroll
  for (int nt = 0; nt < 8; ++nt) {
    int n = nt * 16 + mlane;
    float gg = g[n], bb = bt[n];
#pragma unroll
    for (int r = 0; r < 8; ++r)
      stg[(r + hi16 * 8) * stride + n] = (_Float16)((acc[nt][r] - mean[r]) * inv[r] * gg + bb);
  }
}

// =====================================================================
// FFT front-end (radix-2 DIT 64-pt in LDS, 4 lines / 256-thread block)
// =====================================================================
__device__ __forceinline__ int bitrev6(int x) {
  int r = 0;
#pragma unroll
  for (int i = 0; i < 6; ++i) { r = (r << 1) | (x & 1); x >>= 1; }
  return r;
}

__device__ __forceinline__ void fft64_stages(float2* X, int i) {
#pragma unroll
  for (int s = 1; s <= 6; ++s) {
    int m = 1 << s, hm = m >> 1;
    int j = i & (m - 1);
    int base = i - j;
    float2 res;
    if (j < hm) {
      float2 a = X[base + j], b = X[base + j + hm];
      float ang = -6.283185307179586f * (float)j / (float)m;
      float sn, cs; __sincosf(ang, &sn, &cs);
      res.x = a.x + (cs * b.x - sn * b.y);
      res.y = a.y + (cs * b.y + sn * b.x);
    } else {
      int j2 = j - hm;
      float2 a = X[base + j2], b = X[base + j];
      float ang = -6.283185307179586f * (float)j2 / (float)m;
      float sn, cs; __sincosf(ang, &sn, &cs);
      res.x = a.x - (cs * b.x - sn * b.y);
      res.y = a.y - (cs * b.y + sn * b.x);
    }
    __syncthreads();
    X[i] = res;
    __syncthreads();
  }
}

__device__ __forceinline__ float freqv(int i) { return (i < 32 ? i : i - 64) * (1.f / 64.f); }

// pass 1: FFT along W (contiguous); keep half-spectrum w=0..32
__global__ void __launch_bounds__(256) k_fft_w(const float* __restrict__ src,
                                               float2* __restrict__ dst) {
  __shared__ float2 sh[4][64];
  int line = threadIdx.x >> 6, i = threadIdx.x & 63;
  int lineId = blockIdx.x * 4 + line;                 // over B*C*D*H = 262144 lines
  const float* p = src + (long)lineId * 64;
  sh[line][bitrev6(i)] = make_float2(p[i], 0.f);
  __syncthreads();
  fft64_stages(&sh[line][0], i);
  if (i < WFH) dst[lineId * WFH + i] = sh[line][i];
}

// pass 2: FFT along H (stride 33 float2), in place
__global__ void __launch_bounds__(256) k_fft_h(float2* __restrict__ buf) {
  __shared__ float2 sh[4][64];
  int line = threadIdx.x >> 6, i = threadIdx.x & 63;
  int lineId = blockIdx.x * 4 + line;                 // over (B*C*D)*33 = 135168 lines
  int t = lineId / WFH, w = lineId % WFH;             // t indexes (b,c,d)
  int base = t * (DD * WFH) + w;
  sh[line][bitrev6(i)] = buf[base + i * WFH];
  __syncthreads();
  fft64_stages(&sh[line][0], i);
  buf[base + i * WFH] = sh[line][i];
}

// pass 3: FFT along D + fused |X| radial-band reduction
__global__ void __launch_bounds__(256) k_fft_d(const float2* __restrict__ buf,
                                               const float* __restrict__ edges,
                                               float* __restrict__ num) {
  __shared__ float2 sh[4][64];
  __shared__ float part[4];
  __shared__ float e[5];
  int tid = threadIdx.x;
  int line = tid >> 6, i = tid & 63;
  if (tid < 4) part[tid] = 0.f;
  if (tid < 5) e[tid] = edges[tid];
  int lineId = blockIdx.x * 4 + line;                 // over (B*C)*(H*33) = 135168 lines
  int bc = lineId / (DD * WFH);
  int rem = lineId % (DD * WFH);
  int h = rem / WFH, w = rem % WFH;
  int base = bc * (DD * DD * WFH) + h * WFH + w;
  sh[line][bitrev6(i)] = buf[base + i * (DD * WFH)];
  __syncthreads();
  fft64_stages(&sh[line][0], i);
  float2 X = sh[line][i];
  float mag = sqrtf(X.x * X.x + X.y * X.y);
  float fd = freqv(i), fh = freqv(h), fw = w * (1.f / 64.f);
  float r = sqrtf(fd * fd + fh * fh + fw * fw) * 1.1547005383792515f;  // / sqrt(0.75)
  r = fminf(fmaxf(r, 0.f), 1.f);
#pragma unroll
  for (int k = 0; k < 4; ++k)
    if (r >= e[k] && r < e[k + 1]) atomicAdd(&part[k], mag);
  __syncthreads();
  if (tid < 4) {
    int b = bc >> 5, c = bc & 31;
    atomicAdd(&num[(b * 4 + tid) * CC + c], part[tid]);
  }
}

// mask element counts per band
__global__ void __launch_bounds__(256) k_den(const float* __restrict__ edges,
                                             float* __restrict__ den) {
  __shared__ float part[4];
  __shared__ float e[5];
  int tid = threadIdx.x;
  if (tid < 4) part[tid] = 0.f;
  if (tid < 5) e[tid] = edges[tid];
  __syncthreads();
  int idx = blockIdx.x * 256 + tid;                   // over 64*64*33 = 135168
  if (idx < DD * DD * WFH) {
    int d = idx / (DD * WFH), rem = idx % (DD * WFH);
    int h = rem / WFH, w = rem % WFH;
    float fd = freqv(d), fh = freqv(h), fw = w * (1.f / 64.f);
    float r = sqrtf(fd * fd + fh * fh + fw * fw) * 1.1547005383792515f;
    r = fminf(fmaxf(r, 0.f), 1.f);
#pragma unroll
    for (int k = 0; k < 4; ++k)
      if (r >= e[k] && r < e[k + 1]) atomicAdd(&part[k], 1.f);
  }
  __syncthreads();
  if (tid < 4) atomicAdd(&den[tid], part[tid]);
}

// tiny token chain: tok = num/max(den,eps); fd_tokens = tok@fd_w+b; k/v tokens
__global__ void __launch_bounds__(128) k_tokens(const float* __restrict__ num,
                                                const float* __restrict__ den,
                                                const float* __restrict__ fdw,
                                                const float* __restrict__ fdb,
                                                const float* __restrict__ kw,
                                                const float* __restrict__ kb,
                                                const float* __restrict__ vw,
                                                const float* __restrict__ vb,
                                                float* __restrict__ ktok,
                                                float* __restrict__ vtok) {
  __shared__ float tok[8][32];
  __shared__ float fdt[8][128];
  int t = threadIdx.x;
  if (t < 32)
    for (int row = 0; row < 8; ++row) {
      float d = fmaxf(den[row & 3], 1e-6f);
      tok[row][t] = num[row * 32 + t] / d;
    }
  __syncthreads();
  for (int row = 0; row < 8; ++row) {
    float a = fdb[t];
    for (int c = 0; c < 32; ++c) a += tok[row][c] * fdw[c * 128 + t];
    fdt[row][t] = a;
  }
  __syncthreads();
  for (int row = 0; row < 8; ++row) {
    float ka = kb[t], va = vb[t];
    for (int j = 0; j < 128; ++j) {
      float f = fdt[row][j];
      ka += f * kw[j * 128 + t];
      va += f * vw[j * 128 + t];
    }
    ktok[row * 128 + t] = ka;
    vtok[row * 128 + t] = va;
  }
}

// f32 [K][N] -> f16 transposed [N][K]
__global__ void __launch_bounds__(256) k_cvt(const float* __restrict__ src,
                                             _Float16* __restrict__ dst, int K, int N) {
  int idx = blockIdx.x * 256 + threadIdx.x;
  if (idx < K * N) {
    int k = idx / N, n = idx % N;
    dst[n * K + k] = (_Float16)src[idx];
  }
}

// =====================================================================
// fused spatial pipeline: sp-proj -> LN -> q -> attn(K=4) -> o -> +res
//   -> LN -> MLP(gelu, two 128-wide halves) -> +res -> out-proj + res-proj
// weights resident in LDS (TDM-loaded), WMMA f16->f32 for all GEMMs
// =====================================================================
__global__ void __launch_bounds__(256) k_main(
    const float* __restrict__ sp, const _Float16* __restrict__ wtg,
    const float* __restrict__ sp_b, const float* __restrict__ q_b,
    const float* __restrict__ o_b, const float* __restrict__ m1_b,
    const float* __restrict__ m2_b, const float* __restrict__ op_b,
    const float* __restrict__ r_b, const float* __restrict__ nq_g,
    const float* __restrict__ nq_b, const float* __restrict__ no_g,
    const float* __restrict__ no_b, const float* __restrict__ ktokg,
    const float* __restrict__ vtokg, float* __restrict__ out) {
  extern __shared__ __align__(128) char smem[];
  _Float16* wlds = (_Float16*)smem;                      // WT_TOT halfs
  float* cv = (float*)(wlds + WT_TOT);                   // CV_TOT floats
  _Float16* stageAll = (_Float16*)(cv + CV_TOT);         // 8 * 16*256 halfs
  _Float16* stage2All = stageAll + 8 * 16 * 256;         // 8 * 16*32 halfs

  int tid = threadIdx.x, wave = tid >> 5, lane = tid & 31;
  int mlane = lane & 15, hi16 = lane >> 4, kb = hi16 * 8;
  int tilesPerB = SS / 128;
  int b = blockIdx.x / tilesPerB;
  int s0 = (blockIdx.x % tilesPerB) * 128 + wave * 16;

  // ---- weight table -> LDS: Tensor Data Mover (async, EXEC-independent) ----
#if USE_TDM
  if (wave == 0) {
    unsigned long long ga = (unsigned long long)(uintptr_t)wtg;
    unsigned lad = (unsigned)(uintptr_t)(void*)wlds;
    // D# group0: count=1 | lds_addr | global_addr[56:0] | type=2
    v4u g0 = { 1u, lad, (unsigned)ga,
               (unsigned)((ga >> 32) & 0x01FFFFFFu) | (2u << 30) };
    // D# group1: data_size=8B; tensor 14080x2, tile 14080x2, dim0_stride=14080
    v8i g1 = { 0x00030000, 0x37000000, 0x00020000, 0x37000000, 2, 14080, 0, 0 };
    v4i gz = { 0, 0, 0, 0 };
#if defined(__clang_major__) && __clang_major__ >= 23
    v8i gz8 = { 0, 0, 0, 0, 0, 0, 0, 0 };
    __builtin_amdgcn_tensor_load_to_lds(g0, g1, gz, gz, gz8, 0);
#else
    __builtin_amdgcn_tensor_load_to_lds(g0, g1, gz, gz, 0);
#endif
  }
#else
  {
    const uint4* gs = (const uint4*)wtg;
    uint4* gd = (uint4*)wlds;
    for (int i = tid; i < WT_TOT * 2 / 16; i += 256) gd[i] = gs[i];
  }
#endif

  // ---- small constants (overlap with the TDM transfer) ----
  for (int i = tid; i < 128; i += 256) cv[CV_SPB + i] = sp_b[i];
  for (int i = tid; i < 128; i += 256) cv[CV_QB + i] = q_b[i];
  for (int i = tid; i < 128; i += 256) cv[CV_OB + i] = o_b[i];
  for (int i = tid; i < 256; i += 256) cv[CV_M1B + i] = m1_b[i];
  for (int i = tid; i < 128; i += 256) cv[CV_M2B + i] = m2_b[i];
  for (int i = tid; i < 64; i += 256) cv[CV_OPB + i] = op_b[i];
  for (int i = tid; i < 64; i += 256) cv[CV_RB + i] = r_b[i];
  for (int i = tid; i < 128; i += 256) cv[CV_NQG + i] = nq_g[i];
  for (int i = tid; i < 128; i += 256) cv[CV_NQB + i] = nq_b[i];
  for (int i = tid; i < 128; i += 256) cv[CV_NOG + i] = no_g[i];
  for (int i = tid; i < 128; i += 256) cv[CV_NOB + i] = no_b[i];
  for (int i = tid; i < 512; i += 256) cv[CV_KT + i] = ktokg[b * 512 + i];
  for (int i = tid; i < 512; i += 256) cv[CV_VT + i] = vtokg[b * 512 + i];

#if USE_TDM
  if (wave == 0) {
#if __has_builtin(__builtin_amdgcn_s_wait_tensorcnt)
    __builtin_amdgcn_s_wait_tensorcnt(0);
#else
    asm volatile("s_wait_tensorcnt 0" ::: "memory");
#endif
  }
#endif
  __syncthreads();

  _Float16* stA = stageAll + wave * 16 * 256;            // 16x128 staging
  _Float16* stB = stA + 16 * 128;                        // 16x128 staging
  _Float16* stage2 = stage2All + wave * 16 * 32;         // raw 16x32 input

  // ---- load 16 spatial rows x 32 channels (channel-major global) ----
  const float* spb = sp + (long)b * CC * SS;
  for (int idx = lane; idx < 512; idx += 32) {
    int m = idx & 15, c = idx >> 4;
    stage2[m * 32 + c] = (_Float16)spb[(long)c * SS + s0 + m];
  }
  lds_fence();

  // ---- sp_tokens = X @ sp_w + sp_b ----
  v8f spacc[8];
  gemm_tile<8, 1, false, true>(stage2, 32, wlds + WT_SP, 32, cv + CV_SPB, spacc, mlane, kb, hi16);

  // ---- q = LN(sp_tokens) @ q_w + q_b ----
  ln_dump(spacc, cv + CV_NQG, cv + CV_NQB, stA, 128, mlane, hi16);
  lds_fence();
  v8f qacc[8];
  gemm_tile<8, 4, false, true>(stA, 128, wlds + WT_Q, 128, cv + CV_QB, qacc, mlane, kb, hi16);

  // ---- cross attention with 4 band tokens, 4 heads x 32 dims (VALU) ----
  {
    const float* kt = cv + CV_KT;
    const float* vt = cv + CV_VT;
    const float scale = 0.17677669529663687f;  // 32^-0.5
#pragma unroll
    for (int h = 0; h < 4; ++h) {
      int f0 = h * 32 + mlane, f1 = f0 + 16;
      float sc[8][4];
#pragma unroll
      for (int kk = 0; kk < 4; ++kk) {
        const float* kr = kt + kk * 128;
        float k0 = kr[f0], k1 = kr[f1];
#pragma unroll
        for (int r = 0; r < 8; ++r) {
          float p = qacc[2 * h][r] * k0 + qacc[2 * h + 1][r] * k1;
          sc[r][kk] = red16(p) * scale;
        }
      }
#pragma unroll
      for (int r = 0; r < 8; ++r) {
        float mx = fmaxf(fmaxf(sc[r][0], sc[r][1]), fmaxf(sc[r][2], sc[r][3]));
        float e0 = __expf(sc[r][0] - mx), e1 = __expf(sc[r][1] - mx);
        float e2 = __expf(sc[r][2] - mx), e3 = __expf(sc[r][3] - mx);
        float inv = 1.f / (e0 + e1 + e2 + e3);
        sc[r][0] = e0 * inv; sc[r][1] = e1 * inv; sc[r][2] = e2 * inv; sc[r][3] = e3 * inv;
      }
#pragma unroll
      for (int r = 0; r < 8; ++r) {
        float o0 = 0.f, o1 = 0.f;
#pragma unroll
        for (int kk = 0; kk < 4; ++kk) {
          const float* vr = vt + kk * 128;
          o0 += sc[r][kk] * vr[f0];
          o1 += sc[r][kk] * vr[f1];
        }
        qacc[2 * h][r] = o0;      // reuse q registers as attention output
        qacc[2 * h + 1][r] = o1;
      }
    }
  }

  // ---- attn_out = o @ o_w + o_b ; x = sp_tokens + attn_out ----
  dump_acc(stB, 128, qacc, 8, mlane, hi16);
  lds_fence();
  v8f att[8];
  gemm_tile<8, 4, false, true>(stB, 128, wlds + WT_O, 128, cv + CV_OB, att, mlane, kb, hi16);
#pragma unroll
  for (int nt = 0; nt < 8; ++nt) spacc[nt] = spacc[nt] + att[nt];

  // ---- MLP in two 128-wide halves (keeps wave under 256 VGPRs) ----
  ln_dump(spacc, cv + CV_NOG, cv + CV_NOB, stA, 128, mlane, hi16);
  lds_fence();
#pragma unroll
  for (int hh = 0; hh < 2; ++hh) {
    v8f hacc[8];
    gemm_tile<8, 4, false, true>(stA, 128, wlds + WT_M1 + hh * 128 * 128, 128,
                                 cv + CV_M1B + hh * 128, hacc, mlane, kb, hi16);
#pragma unroll
    for (int nt = 0; nt < 8; ++nt)
#pragma unroll
      for (int i = 0; i < 8; ++i) {
        float x = hacc[nt][i];
        hacc[nt][i] = 0.5f * x * (1.f + erff(x * 0.7071067811865475f));
      }
    dump_acc(stB, 128, hacc, 8, mlane, hi16);
    lds_fence();
    // x += h_half @ m2[K-half, :]   (K window selected by +hh*128 into [N][256] rows)
    gemm_tile<8, 4, true, false>(stB, 128, wlds + WT_M2 + hh * 128, 256,
                                 nullptr, spacc, mlane, kb, hi16);
  }
#pragma unroll
  for (int nt = 0; nt < 8; ++nt) {
    float bv = cv[CV_M2B + nt * 16 + mlane];
#pragma unroll
    for (int i = 0; i < 8; ++i) spacc[nt][i] += bv;
  }

  // ---- out = x @ op_w + op_b + raw @ r_w + r_b ----
  dump_acc(stA, 128, spacc, 8, mlane, hi16);
  lds_fence();
  v8f oacc[4];
  gemm_tile<4, 4, false, true>(stA, 128, wlds + WT_OP, 128, cv + CV_OPB, oacc, mlane, kb, hi16);
  gemm_tile<4, 1, true, false>(stage2, 32, wlds + WT_R, 32, nullptr, oacc, mlane, kb, hi16);
#pragma unroll
  for (int nt = 0; nt < 4; ++nt) {
    float bv = cv[CV_RB + nt * 16 + mlane];
#pragma unroll
    for (int i = 0; i < 8; ++i) oacc[nt][i] += bv;
  }

  // ---- transpose in LDS, then coalesced channel-major store ----
  float* stf = (float*)stA;  // 16x64 f32 fits in the 4KB half-stage
#pragma unroll
  for (int nt = 0; nt < 4; ++nt)
#pragma unroll
    for (int r = 0; r < 8; ++r)
      stf[(r + hi16 * 8) * 64 + nt * 16 + mlane] = oacc[nt][r];
  lds_fence();
  float* outb = out + (long)b * 64 * SS;
  for (int idx = lane; idx < 1024; idx += 32) {
    int m = idx & 15, oc = idx >> 4;
    outb[(long)oc * SS + s0 + m] = stf[m * 64 + oc];
  }
}

// =====================================================================
// launch
// =====================================================================
extern "C" void kernel_launch(void* const* d_in, const int* in_sizes, int n_in,
                              void* d_out, int out_size, void* d_ws, size_t ws_size,
                              hipStream_t stream) {
  (void)in_sizes; (void)n_in; (void)out_size; (void)ws_size;
  const float* fd_feat = (const float*)d_in[0];
  const float* sp_feat = (const float*)d_in[1];
  const float* edges   = (const float*)d_in[2];
  const float* fd_w = (const float*)d_in[3];
  const float* fd_b = (const float*)d_in[4];
  const float* sp_w = (const float*)d_in[5];
  const float* sp_b = (const float*)d_in[6];
  const float* q_w  = (const float*)d_in[7];
  const float* q_b  = (const float*)d_in[8];
  const float* k_w  = (const float*)d_in[9];
  const float* k_b  = (const float*)d_in[10];
  const float* v_w  = (const float*)d_in[11];
  const float* v_b  = (const float*)d_in[12];
  const float* o_w  = (const float*)d_in[13];
  const float* o_b  = (const float*)d_in[14];
  const float* nq_g = (const float*)d_in[15];
  const float* nq_b = (const float*)d_in[16];
  const float* no_g = (const float*)d_in[17];
  const float* no_b = (const float*)d_in[18];
  const float* m1_w = (const float*)d_in[19];
  const float* m1_b = (const float*)d_in[20];
  const float* m2_w = (const float*)d_in[21];
  const float* m2_b = (const float*)d_in[22];
  const float* op_w = (const float*)d_in[23];
  const float* op_b = (const float*)d_in[24];
  const float* r_w  = (const float*)d_in[25];
  const float* r_b  = (const float*)d_in[26];

  char* ws = (char*)d_ws;
  float2* fftb = (float2*)(ws + OFF_FFT);
  float* num = (float*)(ws + OFF_NUM);
  float* den = (float*)(ws + OFF_DEN);
  float* ktok = (float*)(ws + OFF_KT);
  float* vtok = (float*)(ws + OFF_VT);
  _Float16* wt = (_Float16*)(ws + OFF_WT);

  hipMemsetAsync(ws + OFF_NUM, 0, (size_t)(OFF_KT - OFF_NUM), stream);

  // spectral front-end
  k_fft_w<<<65536, 256, 0, stream>>>(fd_feat, fftb);
  k_fft_h<<<33792, 256, 0, stream>>>(fftb);
  k_fft_d<<<33792, 256, 0, stream>>>(fftb, edges, num);
  k_den<<<528, 256, 0, stream>>>(edges, den);
  k_tokens<<<1, 128, 0, stream>>>(num, den, fd_w, fd_b, k_w, k_b, v_w, v_b, ktok, vtok);

  // weight conversion (f32 [K][N] -> f16 [N][K])
  k_cvt<<<16, 256, 0, stream>>>(sp_w, wt + WT_SP, 32, 128);
  k_cvt<<<64, 256, 0, stream>>>(q_w, wt + WT_Q, 128, 128);
  k_cvt<<<64, 256, 0, stream>>>(o_w, wt + WT_O, 128, 128);
  k_cvt<<<128, 256, 0, stream>>>(m1_w, wt + WT_M1, 128, 256);
  k_cvt<<<128, 256, 0, stream>>>(m2_w, wt + WT_M2, 256, 128);
  k_cvt<<<32, 256, 0, stream>>>(op_w, wt + WT_OP, 128, 64);
  k_cvt<<<8, 256, 0, stream>>>(r_w, wt + WT_R, 32, 64);

  // fused WMMA pipeline: 2 batches * 2048 tiles of 128 rows
  k_main<<<BB * (SS / 128), 256, LDS_MAIN_BYTES, stream>>>(
      sp_feat, wt, sp_b, q_b, o_b, m1_b, m2_b, op_b, r_b,
      nq_g, nq_b, no_g, no_b, ktok, vtok, (float*)d_out);
}
